// GCHALayer_25701084299355
// MI455X (gfx1250) — matrix operand order repaired and checked by reading, not compiled, not run on hardware
//
#include <hip/hip_runtime.h>

// ---------------------------------------------------------------------------
// GCHA layer for MI455X (gfx1250, wave32).
// All matmuls use v_wmma_f32_16x16x32_bf16 (f32 accumulate, bf16 operands).
// attn [B,H,N,HW] f32 (512MB) is written exactly once (HBM roofline ~22us).
// Softmax: THREE cheap sweeps instead of one expensive online sweep:
//   (1) unmasked row max  -- no mask, no selects, 1 VALU/elem (max is only a
//       numerical stabilizer, so including masked scores is harmless),
//   (2) masked sum of exp2(s - M) -- no overflow since s <= M; lane partials
//       merge by plain addition (common M),
//   (3) p = exp2(s - (M + log2(S))) masked; write attn + ctx WMMAs.
// Base-2 exponentials with 1/sqrt(DK)*log2(e) folded into the Q projection.
// Geometric mask is precomputed as a 512KB bitmask.
// ---------------------------------------------------------------------------

#define B_  4
#define N_  1024
#define HW_ 4096
#define D_  512
#define H_  8
#define DK_ 64
#define NEG_BIG (-1.0e30f)
#define EXP2(x) __builtin_amdgcn_exp2f(x)
#define LOG2(x) __builtin_amdgcn_logf(x)
// log2(e) / sqrt(DK) = 1.4426950408889634 / 8
#define QSCALE 0.18033688011112043f

typedef __attribute__((ext_vector_type(16))) __bf16 bfx16;
typedef __attribute__((ext_vector_type(8)))  __bf16 bfx8;
typedef __attribute__((ext_vector_type(8)))  float  fx8;

static __device__ __forceinline__ fx8 wmma_bf16(bfx16 a, bfx16 b, fx8 c) {
  // 8 args: (neg_a, A, neg_b, B, c_mod, C, reuse_a, reuse_b)
  return __builtin_amdgcn_wmma_f32_16x16x32_bf16(false, a, false, b,
                                                 (short)0, c, false, false);
}

// convert 8 consecutive f32 -> 8 bf16 (two b128 loads)
static __device__ __forceinline__ bfx8 cvt8(const float* p) {
  float4 a = *(const float4*)(p);
  float4 b = *(const float4*)(p + 4);
  bfx8 r;
  r[0] = (__bf16)a.x; r[1] = (__bf16)a.y; r[2] = (__bf16)a.z; r[3] = (__bf16)a.w;
  r[4] = (__bf16)b.x; r[5] = (__bf16)b.y; r[6] = (__bf16)b.z; r[7] = (__bf16)b.w;
  return r;
}

// A-operand (16x32, MxK): lane l holds M=l%16; elems 0-7 = K(8*(l/16)..+7),
// elems 8-15 = K(16+8*(l/16)..+7).  Row-major source, contiguous along K.
static __device__ __forceinline__ bfx16 fragA_f32(const float* base, int ld, int col0) {
  const int lane = threadIdx.x & 31;
  const float* rp = base + (size_t)(lane & 15) * ld + col0 + 8 * (lane >> 4);
  union { bfx16 v; bfx8 h[2]; } u;
  u.h[0] = cvt8(rp);
  u.h[1] = cvt8(rp + 16);
  return u.v;
}
static __device__ __forceinline__ bfx16 fragA_bf16(const __bf16* base, int ld, int col0) {
  const int lane = threadIdx.x & 31;
  const __bf16* rp = base + (size_t)(lane & 15) * ld + col0 + 8 * (lane >> 4);
  union { bfx16 v; bfx8 h[2]; } u;
  u.h[0] = *(const bfx8*)(rp);
  u.h[1] = *(const bfx8*)(rp + 16);
  return u.v;
}
// B-operand (32x16, KxN): lane l holds N=l%16; elems 0-7 = K(8*(l/16)..+7),
// elems 8-15 = K(16+8*(l/16)..+7) (same K order as the A-operand).
// Source is row-major with rows = N, contiguous along K.
static __device__ __forceinline__ bfx16 fragB_f32(const float* base, int ld, int col0) {
  const int lane = threadIdx.x & 31;
  const float* rp = base + (size_t)(lane & 15) * ld + col0 + 8 * (lane >> 4);
  union { bfx16 v; bfx8 h[2]; } u;
  u.h[0] = cvt8(rp);
  u.h[1] = cvt8(rp + 16);
  return u.v;
}
static __device__ __forceinline__ bfx16 fragB_bf16(const __bf16* base, int ld, int col0) {
  const int lane = threadIdx.x & 31;
  const __bf16* rp = base + (size_t)(lane & 15) * ld + col0 + 8 * (lane >> 4);
  union { bfx16 v; bfx8 h[2]; } u;
  u.h[0] = *(const bfx8*)(rp);
  u.h[1] = *(const bfx8*)(rp + 16);
  return u.v;
}

// ---------------------------------------------------------------------------
// Kernel 1: geometric mask bitmask.  bit(n,m) = ||x_tilde[m]-x_hat[n,m]|| < eps
// ---------------------------------------------------------------------------
__global__ __launch_bounds__(256)
void mask_kernel(const float* __restrict__ xt, const float* __restrict__ xh,
                 unsigned* __restrict__ msk) {
  const int t = blockIdx.x * 256 + threadIdx.x;   // t = n*HW + m
  const int m = t & (HW_ - 1);
  const float dx = xt[2 * m]     - xh[2 * (size_t)t];
  const float dy = xt[2 * m + 1] - xh[2 * (size_t)t + 1];
  const bool inside = (dx * dx + dy * dy) < 0.01f;  // eps^2, eps = 0.1
  const unsigned long long bal = __ballot(inside);  // wave32: low 32 bits
  if ((threadIdx.x & 31) == 0) msk[t >> 5] = (unsigned)bal;
}

// ---------------------------------------------------------------------------
// Kernel 2: C = alpha * (A @ W^T + bias).  A:[M,512] f32, W:[512,512] f32.
// MODE 0: store bf16 row-major [M,512]        (Q with alpha=QSCALE, K)
// MODE 1: store bf16 transposed per batch: out[b][d][m], m=row%HW  (V^T)
// MODE 2: store f32 row-major [M,512]         (final output)
// Block = 128 threads (4 waves); each wave computes a 32x64 tile.
// ---------------------------------------------------------------------------
template <int MODE>
__global__ __launch_bounds__(128)
void proj_kernel(const float* __restrict__ A, const float* __restrict__ W,
                 const float* __restrict__ bias, void* __restrict__ out,
                 float alpha) {
  const int lane = threadIdx.x & 31;
  const int wv = threadIdx.x >> 5;
  const int rowBase = blockIdx.x * 128 + wv * 32;
  const int colBase = blockIdx.y * 64;

  fx8 acc[2][4];
#pragma unroll
  for (int i = 0; i < 2; ++i)
#pragma unroll
    for (int j = 0; j < 4; ++j) acc[i][j] = (fx8){0.f,0.f,0.f,0.f,0.f,0.f,0.f,0.f};

  for (int k0 = 0; k0 < D_; k0 += 32) {
    bfx16 a0 = fragA_f32(A + (size_t)rowBase * D_, D_, k0);
    bfx16 a1 = fragA_f32(A + (size_t)(rowBase + 16) * D_, D_, k0);
#pragma unroll
    for (int j = 0; j < 4; ++j) {
      bfx16 bj = fragB_f32(W + (size_t)(colBase + 16 * j) * D_, D_, k0);
      acc[0][j] = wmma_bf16(a0, bj, acc[0][j]);
      acc[1][j] = wmma_bf16(a1, bj, acc[1][j]);
    }
  }

#pragma unroll
  for (int i = 0; i < 2; ++i)
#pragma unroll
    for (int j = 0; j < 4; ++j) {
      const int col = colBase + 16 * j + (lane & 15);
      const float bv = bias[col];
#pragma unroll
      for (int r = 0; r < 8; ++r) {
        const int row = rowBase + 16 * i + r + 8 * (lane >> 4);
        const float v = (acc[i][j][r] + bv) * alpha;
        if (MODE == 0) {
          ((__bf16*)out)[(size_t)row * D_ + col] = (__bf16)v;
        } else if (MODE == 1) {
          const int bb = row >> 12;           // row / HW  (HW = 4096)
          const int m = row & (HW_ - 1);
          ((__bf16*)out)[((size_t)bb * D_ + col) * HW_ + m] = (__bf16)v;
        } else {
          ((float*)out)[(size_t)row * D_ + col] = v;
        }
      }
    }
}

// ---------------------------------------------------------------------------
// Kernel 3: fused attention.  One wave = 16 query rows of one (b,h).
// Scores arrive pre-scaled to the log2 domain (QSCALE folded into Q).
// ---------------------------------------------------------------------------
__global__ __launch_bounds__(128)
void attn_kernel(const void* __restrict__ Qb_, const void* __restrict__ Kb_,
                 const void* __restrict__ Vt_, const unsigned* __restrict__ msk,
                 float* __restrict__ attn, float* __restrict__ ctx) {
  const __bf16* Qb = (const __bf16*)Qb_;
  const __bf16* Kb = (const __bf16*)Kb_;
  const __bf16* Vt = (const __bf16*)Vt_;

  __shared__ __align__(16) __bf16 pls[4][16][40];  // per-wave P tile, padded

  const int lane = threadIdx.x & 31;
  const int wv = threadIdx.x >> 5;
  const int wid = blockIdx.x * 4 + wv;
  const int qt = wid & 63;          // N/16 = 64 query tiles
  const int h = (wid >> 6) & 7;
  const int b = wid >> 9;
  const int n0 = qt * 16;
  const int lh = lane >> 4;         // lane half (0/1)
  const int ll = lane & 15;

  const __bf16* Qp = Qb + ((size_t)b * N_ + n0) * D_ + h * DK_;
  const __bf16* Kp = Kb + (size_t)b * HW_ * D_ + h * DK_;
  const __bf16* Vp = Vt + ((size_t)b * D_ + h * DK_) * HW_;

  const bfx16 aq0 = fragA_bf16(Qp, D_, 0);
  const bfx16 aq1 = fragA_bf16(Qp, D_, 32);

  const unsigned bit0 = 1u << ll;          // sub-tile 0 bit in the mask word
  const unsigned bit1 = bit0 << 16;        // sub-tile 1 bit

  // ---- pass 1: UNMASKED row max (numerical stabilizer only) ---------------
  // No mask loads, no selects: masked scores can only raise M, which is safe.
  fx8 vmax = (fx8){NEG_BIG,NEG_BIG,NEG_BIG,NEG_BIG,NEG_BIG,NEG_BIG,NEG_BIG,NEG_BIG};
  for (int m0 = 0; m0 < HW_; m0 += 32) {
    fx8 s0 = (fx8){0.f,0.f,0.f,0.f,0.f,0.f,0.f,0.f};
    fx8 s1 = s0;
    const bfx16 k00 = fragB_bf16(Kp + (size_t)m0 * D_, D_, 0);
    const bfx16 k01 = fragB_bf16(Kp + (size_t)m0 * D_, D_, 32);
    s0 = wmma_bf16(aq0, k00, s0);
    s0 = wmma_bf16(aq1, k01, s0);
    const bfx16 k10 = fragB_bf16(Kp + (size_t)(m0 + 16) * D_, D_, 0);
    const bfx16 k11 = fragB_bf16(Kp + (size_t)(m0 + 16) * D_, D_, 32);
    s1 = wmma_bf16(aq0, k10, s1);
    s1 = wmma_bf16(aq1, k11, s1);
#pragma unroll
    for (int r = 0; r < 8; ++r)
      vmax[r] = fmaxf(vmax[r], fmaxf(s0[r], s1[r]));
  }
  float M[8];
#pragma unroll
  for (int r = 0; r < 8; ++r) {
    float m = vmax[r];
#pragma unroll
    for (int off = 1; off <= 8; off <<= 1)
      m = fmaxf(m, __shfl_xor(m, off));   // reduce within 16-lane half
    M[r] = m;
  }

  // ---- pass 2: masked sum of exp2(s - M); lane partials add directly ------
  float lsum[8];
#pragma unroll
  for (int r = 0; r < 8; ++r) lsum[r] = 0.f;
  for (int m0 = 0; m0 < HW_; m0 += 32) {
    fx8 s0 = (fx8){0.f,0.f,0.f,0.f,0.f,0.f,0.f,0.f};
    fx8 s1 = s0;
    const bfx16 k00 = fragB_bf16(Kp + (size_t)m0 * D_, D_, 0);
    const bfx16 k01 = fragB_bf16(Kp + (size_t)m0 * D_, D_, 32);
    s0 = wmma_bf16(aq0, k00, s0);
    s0 = wmma_bf16(aq1, k01, s0);
    const bfx16 k10 = fragB_bf16(Kp + (size_t)(m0 + 16) * D_, D_, 0);
    const bfx16 k11 = fragB_bf16(Kp + (size_t)(m0 + 16) * D_, D_, 32);
    s1 = wmma_bf16(aq0, k10, s1);
    s1 = wmma_bf16(aq1, k11, s1);
    const int mw = m0 >> 5;
#pragma unroll
    for (int r = 0; r < 8; ++r) {
      const int q = n0 + r + 8 * lh;
      const unsigned w = msk[(size_t)q * (HW_ / 32) + mw];
      const float ea = (w & bit0) ? EXP2(s0[r] - M[r]) : 0.f;  // s <= M: no ovfl
      const float eb = (w & bit1) ? EXP2(s1[r] - M[r]) : 0.f;
      lsum[r] += ea + eb;
    }
  }
  float shift[8];
#pragma unroll
  for (int r = 0; r < 8; ++r) {
    float s = lsum[r];
#pragma unroll
    for (int off = 1; off <= 8; off <<= 1)
      s += __shfl_xor(s, off);            // plain add: common M per row
    shift[r] = M[r] + LOG2(s);
  }

  // ---- pass 3: attn write + ctx accumulate ---------------------------------
  fx8 c[4];
#pragma unroll
  for (int d = 0; d < 4; ++d) c[d] = (fx8){0.f,0.f,0.f,0.f,0.f,0.f,0.f,0.f};

  for (int m0 = 0; m0 < HW_; m0 += 32) {
    fx8 s0 = (fx8){0.f,0.f,0.f,0.f,0.f,0.f,0.f,0.f};
    fx8 s1 = s0;
    const bfx16 k00 = fragB_bf16(Kp + (size_t)m0 * D_, D_, 0);
    const bfx16 k01 = fragB_bf16(Kp + (size_t)m0 * D_, D_, 32);
    s0 = wmma_bf16(aq0, k00, s0);
    s0 = wmma_bf16(aq1, k01, s0);
    const bfx16 k10 = fragB_bf16(Kp + (size_t)(m0 + 16) * D_, D_, 0);
    const bfx16 k11 = fragB_bf16(Kp + (size_t)(m0 + 16) * D_, D_, 32);
    s1 = wmma_bf16(aq0, k10, s1);
    s1 = wmma_bf16(aq1, k11, s1);
    const int mw = m0 >> 5;
#pragma unroll
    for (int r = 0; r < 8; ++r) {
      const int q = n0 + r + 8 * lh;
      const unsigned w = msk[(size_t)q * (HW_ / 32) + mw];
      float* arow = attn + ((size_t)(b * H_ + h) * N_ + q) * HW_ + m0 + ll;
      const float pa = (w & bit0) ? EXP2(s0[r] - shift[r]) : 0.f;
      arow[0] = pa;
      pls[wv][r + 8 * lh][ll] = (__bf16)pa;
      const float pb = (w & bit1) ? EXP2(s1[r] - shift[r]) : 0.f;
      arow[16] = pb;
      pls[wv][r + 8 * lh][16 + ll] = (__bf16)pb;
    }
    // same-wave LDS ops are in-order; just stop compiler reordering.
    asm volatile("" ::: "memory");
    union { bfx16 v; bfx8 hh[2]; } up;   // read back in A-operand layout
    const __bf16* pr = &pls[wv][ll][0];
    up.hh[0] = *(const bfx8*)(pr + 8 * lh);
    up.hh[1] = *(const bfx8*)(pr + 16 + 8 * lh);
#pragma unroll
    for (int d = 0; d < 4; ++d) {
      const bfx16 bv = fragB_bf16(Vp + (size_t)(16 * d) * HW_ + m0, HW_, 0);
      c[d] = wmma_bf16(up.v, bv, c[d]);
    }
    asm volatile("" ::: "memory");
  }

#pragma unroll
  for (int d = 0; d < 4; ++d)
#pragma unroll
    for (int r = 0; r < 8; ++r)
      ctx[((size_t)b * N_ + n0 + r + 8 * lh) * D_ + h * DK_ + 16 * d + ll] = c[d][r];
}

// ---------------------------------------------------------------------------
extern "C" void kernel_launch(void* const* d_in, const int* in_sizes, int n_in,
                              void* d_out, int out_size, void* d_ws, size_t ws_size,
                              hipStream_t stream) {
  (void)in_sizes; (void)n_in; (void)out_size; (void)ws_size;
  const float* query = (const float*)d_in[0];
  const float* key   = (const float*)d_in[1];
  const float* value = (const float*)d_in[2];
  const float* xt    = (const float*)d_in[3];
  const float* xh    = (const float*)d_in[4];
  const float* Wq = (const float*)d_in[5];  const float* bq = (const float*)d_in[6];
  const float* Wk = (const float*)d_in[7];  const float* bk = (const float*)d_in[8];
  const float* Wv = (const float*)d_in[9];  const float* bv = (const float*)d_in[10];
  const float* Wo = (const float*)d_in[11]; const float* bo = (const float*)d_in[12];

  // workspace carve-out (~45 MB): Qb 4MB | Kb 16MB | Vt 16MB | ctx 8MB | mask 0.5MB
  char* ws = (char*)d_ws;
  void* Qb = (void*)ws;                 ws += (size_t)B_ * N_ * D_ * 2;
  void* Kb = (void*)ws;                 ws += (size_t)B_ * HW_ * D_ * 2;
  void* Vt = (void*)ws;                 ws += (size_t)B_ * HW_ * D_ * 2;
  float* ctx = (float*)ws;              ws += (size_t)B_ * N_ * D_ * 4;
  unsigned* msk = (unsigned*)ws;

  float* outp = (float*)d_out;                       // out [B,N,D]
  float* attn = outp + (size_t)B_ * N_ * D_;         // attn [B,H,N,HW]

  mask_kernel<<<(N_ * HW_) / 256, 256, 0, stream>>>(xt, xh, msk);

  const dim3 blk(128);
  // Q carries the softmax scale in the log2 domain: alpha = log2(e)/sqrt(DK)
  proj_kernel<0><<<dim3((B_ * N_) / 128, D_ / 64),  blk, 0, stream>>>(query, Wq, bq, Qb, (float)QSCALE);
  proj_kernel<0><<<dim3((B_ * HW_) / 128, D_ / 64), blk, 0, stream>>>(key,   Wk, bk, Kb, 1.0f);
  proj_kernel<1><<<dim3((B_ * HW_) / 128, D_ / 64), blk, 0, stream>>>(value, Wv, bv, Vt, 1.0f);

  attn_kernel<<<(B_ * H_ * (N_ / 16)) / 4, blk, 0, stream>>>(Qb, Kb, Vt, msk, attn, ctx);

  proj_kernel<2><<<dim3((B_ * N_) / 128, D_ / 64), blk, 0, stream>>>(ctx, Wo, bo, (void*)outp, 1.0f);
}
